// XrayNet_65859028517272
// MI455X (gfx1250) — matrix operand order repaired and checked by reading, not compiled, not run on hardware
//
#include <hip/hip_runtime.h>
#include <hip/hip_bf16.h>

// ---------------------------------------------------------------------------
// CDNA5 (gfx1250) LSTM captioning decoder.
//   base1 = cnn @ Wih1[:, :1024]^T + b_ih1 + b_hh1      (once, WMMA bf16)
//   embT  = embedding @ Wih1[:, 1024:]^T                (once, WMMA bf16)
//   recurrence: ONE persistent kernel, 513 pipelined phases:
//     phase p: LSTM1(step p)  ||  LSTM2(step p-1)   (share only read-only h1)
//     grid barrier (agent-scope atomics + s_sleep) between phases
//   logits = h2_seq @ Wout^T + b_out -> log_softmax     (WMMA bf16, f32 acc)
// ---------------------------------------------------------------------------

typedef __bf16 bf16_t;
typedef __attribute__((ext_vector_type(16))) __bf16 v16bf;
typedef __attribute__((ext_vector_type(8)))  __bf16 v8bf;
typedef __attribute__((ext_vector_type(8)))  float  v8f;

// ---- WMMA tile loaders (layouts per CDNA5 ISA §7.12.2) ----------

// A: M x K row-major (K contiguous). lane holds row (lane&15);
// lanes 0-15: K = k0+{0..7, 16..23}; lanes 16-31: K = k0+{8..15, 24..31}.
__device__ __forceinline__ v16bf load_a_tile(const bf16_t* A, int lda,
                                             int mbase, int k0, int lane) {
  int m  = mbase + (lane & 15);
  int kg = (lane < 16) ? 0 : 8;
  const bf16_t* p = A + (size_t)m * lda + k0 + kg;
  v8bf lo = *(const v8bf*)(p);        // 16B load: K = kg..kg+7
  v8bf hi = *(const v8bf*)(p + 16);   // 16B load: K = kg+16..kg+23
  v16bf a;
#pragma unroll
  for (int i = 0; i < 8; ++i) { a[i] = lo[i]; a[i + 8] = hi[i]; }
  return a;
}

// B: K x N with W stored N x K row-major (we compute X @ W^T), so
// B element (k,n) = W[n][k]. lane holds column n = lane&15;
// lanes 0-15: K = k0+{0..7, 16..23}; lanes 16-31: K = k0+{8..15, 24..31}.
__device__ __forceinline__ v16bf load_b_tile(const bf16_t* W, int ldb,
                                             int nbase, int k0, int lane) {
  int n  = nbase + (lane & 15);
  int kg = (lane < 16) ? 0 : 8;
  const bf16_t* p = W + (size_t)n * ldb + k0 + kg;
  v8bf lo = *(const v8bf*)(p);
  v8bf hi = *(const v8bf*)(p + 16);
  v16bf b;
#pragma unroll
  for (int i = 0; i < 8; ++i) { b[i] = lo[i]; b[i + 8] = hi[i]; }
  return b;
}

__device__ __forceinline__ v8f wmma_bf16(v16bf a, v16bf b, v8f c) {
  return __builtin_amdgcn_wmma_f32_16x16x32_bf16(
      /*neg_a=*/false, a, /*neg_b=*/false, b,
      /*c_mod=*/(short)0, c, /*reuse_a=*/false, /*reuse_b=*/false);
}

__device__ __forceinline__ float sigmoidf_fast(float x) {
  return 1.0f / (1.0f + __expf(-x));
}

// ---- grid-wide barrier (sense-reversal via generation counter) -------------
__device__ __forceinline__ void grid_barrier(unsigned* cnt, unsigned* gen,
                                             unsigned nblk) {
  __threadfence();           // flush this thread's global stores
  __syncthreads();
  if (threadIdx.x == 0) {
    unsigned g = __hip_atomic_load(gen, __ATOMIC_RELAXED,
                                   __HIP_MEMORY_SCOPE_AGENT);
    unsigned arrived = __hip_atomic_fetch_add(cnt, 1u, __ATOMIC_ACQ_REL,
                                              __HIP_MEMORY_SCOPE_AGENT);
    if (arrived == nblk - 1) {
      __hip_atomic_store(cnt, 0u, __ATOMIC_RELAXED, __HIP_MEMORY_SCOPE_AGENT);
      __hip_atomic_fetch_add(gen, 1u, __ATOMIC_RELEASE,
                             __HIP_MEMORY_SCOPE_AGENT);
    } else {
      while (__hip_atomic_load(gen, __ATOMIC_ACQUIRE,
                               __HIP_MEMORY_SCOPE_AGENT) == g)
        __builtin_amdgcn_s_sleep(1);
    }
  }
  __syncthreads();
  asm volatile("" ::: "memory");  // no global-access reordering across barrier
}

// ---- small utility kernels -------------------------------------------------

__global__ void k_f32_to_bf16(const float* __restrict__ in,
                              bf16_t* __restrict__ out, int n) {
  int i = blockIdx.x * blockDim.x + threadIdx.x;
  int stride = gridDim.x * blockDim.x;
  for (; i < n; i += stride) out[i] = (bf16_t)in[i];
}

__global__ void k_add_bias(const float* __restrict__ a,
                           const float* __restrict__ b,
                           float* __restrict__ o, int n) {
  int i = blockIdx.x * blockDim.x + threadIdx.x;
  if (i < n) o[i] = a[i] + b[i];
}

// ---- generic WMMA GEMM: C[M,N] = A[M,K] @ W[N,K]^T (+bias0+bias1) ----------

__global__ void k_wmma_gemm_bias(const bf16_t* __restrict__ A, int lda,
                                 const bf16_t* __restrict__ W, int ldb,
                                 float* __restrict__ C, int ldc,
                                 int Mtiles, int Ntiles, int K,
                                 const float* __restrict__ bias0,
                                 const float* __restrict__ bias1) {
  int wave = (blockIdx.x * blockDim.x + threadIdx.x) >> 5;
  int lane = threadIdx.x & 31;
  if (wave >= Mtiles * Ntiles) return;  // wave-uniform: EXEC stays all-1s
  int mt = wave / Ntiles, nt = wave % Ntiles;
  v8f acc = {};
  for (int k0 = 0; k0 < K; k0 += 32) {
    v16bf a = load_a_tile(A, lda, mt * 16, k0, lane);
    v16bf b = load_b_tile(W, ldb, nt * 16, k0, lane);
    acc = wmma_bf16(a, b, acc);
  }
  int n    = nt * 16 + (lane & 15);
  int moff = (lane < 16) ? 0 : 8;
  float badd = (bias0 ? bias0[n] : 0.0f) + (bias1 ? bias1[n] : 0.0f);
#pragma unroll
  for (int r = 0; r < 8; ++r) {
    int m = mt * 16 + moff + r;
    C[(size_t)m * ldc + n] = acc[r] + badd;
  }
}

// ---- persistent recurrence kernel ------------------------------------------
// Exactly 128 waves = 32 blocks x 128 threads (no early return: all threads
// reach every barrier). Tiling: 4 M-tiles (batch 64) x 32 J-tiles (hidden
// 512); each wave owns matched i/f/g/o gate columns -> cell math wave-local.
// h1(l) lives in ((l&1) ? h1a : h1b); same pattern for h2. All phase-crossing
// dependencies are separated by exactly one grid barrier.

__global__ void k_lstm_persistent(const int*    __restrict__ gt,     // (64,512)
                                  const float*  __restrict__ base1,  // (64,2048)
                                  const float*  __restrict__ embT,   // (2000,2048)
                                  const float*  __restrict__ base2,  // (2048)
                                  const bf16_t* __restrict__ whh1,   // (2048,512)
                                  const bf16_t* __restrict__ wih2,   // (2048,512)
                                  const bf16_t* __restrict__ whh2,   // (2048,512)
                                  float* c1, float* c2,              // (64,512)
                                  bf16_t* h1a, bf16_t* h1b,
                                  bf16_t* h2a, bf16_t* h2b,
                                  bf16_t* h2_seq,                    // (512,64,512)
                                  unsigned* bar_cnt, unsigned* bar_gen) {
  const int lane = threadIdx.x & 31;
  const int wave = (blockIdx.x * blockDim.x + threadIdx.x) >> 5;  // 0..127
  const int mt   = wave >> 5;                                     // 0..3
  const int jt   = wave & 31;                                     // 0..31
  const int j    = jt * 16 + (lane & 15);
  const int moff = (lane < 16) ? 0 : 8;
  const unsigned nblk = gridDim.x;

  for (int ph = 0; ph <= 512; ++ph) {
    // ======== LSTM1 for step l = ph ========
    if (ph < 512) {
      const int l = ph;
      const bf16_t* h1_in  = (l & 1) ? h1b : h1a;   // h1(l-1)
      bf16_t*       h1_out = (l & 1) ? h1a : h1b;   // h1(l)
      v8f acc[4] = {};
      for (int k0 = 0; k0 < 512; k0 += 32) {
        v16bf a = load_a_tile(h1_in, 512, mt * 16, k0, lane);
#pragma unroll
        for (int g = 0; g < 4; ++g) {
          v16bf b = load_b_tile(whh1, 512, g * 512 + jt * 16, k0, lane);
          acc[g] = wmma_bf16(a, b, acc[g]);
        }
      }
#pragma unroll
      for (int r = 0; r < 8; ++r) {
        int b   = mt * 16 + moff + r;
        int tok = (l == 0) ? 0 : gt[b * 512 + (l - 1)];  // SOS = 0
        const float* bb = base1 + (size_t)b * 2048;
        const float* eb = embT + (size_t)tok * 2048;
        float gi = acc[0][r] + bb[j]        + eb[j];
        float gf = acc[1][r] + bb[512 + j]  + eb[512 + j];
        float gg = acc[2][r] + bb[1024 + j] + eb[1024 + j];
        float go = acc[3][r] + bb[1536 + j] + eb[1536 + j];
        float i_ = sigmoidf_fast(gi);
        float f_ = sigmoidf_fast(gf);
        float g_ = tanhf(gg);
        float o_ = sigmoidf_fast(go);
        float c  = f_ * c1[b * 512 + j] + i_ * g_;
        c1[b * 512 + j]     = c;
        h1_out[b * 512 + j] = (bf16_t)(o_ * tanhf(c));
      }
    }
    // ======== LSTM2 for step s = ph-1 (reads h1(s): written phase ph-1) ====
    if (ph >= 1) {
      const int s = ph - 1;
      const bf16_t* h1s    = (s & 1) ? h1a : h1b;   // h1(s)
      const bf16_t* h2_in  = (s & 1) ? h2b : h2a;   // h2(s-1)
      bf16_t*       h2_out = (s & 1) ? h2a : h2b;   // h2(s)
      v8f acc[4] = {};
      for (int k0 = 0; k0 < 512; k0 += 32) {        // h1 @ Wih2^T
        v16bf a = load_a_tile(h1s, 512, mt * 16, k0, lane);
#pragma unroll
        for (int g = 0; g < 4; ++g) {
          v16bf b = load_b_tile(wih2, 512, g * 512 + jt * 16, k0, lane);
          acc[g] = wmma_bf16(a, b, acc[g]);
        }
      }
      for (int k0 = 0; k0 < 512; k0 += 32) {        // h2 @ Whh2^T
        v16bf a = load_a_tile(h2_in, 512, mt * 16, k0, lane);
#pragma unroll
        for (int g = 0; g < 4; ++g) {
          v16bf b = load_b_tile(whh2, 512, g * 512 + jt * 16, k0, lane);
          acc[g] = wmma_bf16(a, b, acc[g]);
        }
      }
#pragma unroll
      for (int r = 0; r < 8; ++r) {
        int b = mt * 16 + moff + r;
        float gi = acc[0][r] + base2[j];
        float gf = acc[1][r] + base2[512 + j];
        float gg = acc[2][r] + base2[1024 + j];
        float go = acc[3][r] + base2[1536 + j];
        float i_ = sigmoidf_fast(gi);
        float f_ = sigmoidf_fast(gf);
        float g_ = tanhf(gg);
        float o_ = sigmoidf_fast(go);
        float c  = f_ * c2[b * 512 + j] + i_ * g_;
        c2[b * 512 + j] = c;
        bf16_t hb = (bf16_t)(o_ * tanhf(c));
        h2_out[b * 512 + j] = hb;
        h2_seq[((size_t)s * 64 + b) * 512 + j] = hb;
      }
    }
    if (ph < 512) grid_barrier(bar_cnt, bar_gen, nblk);
  }
}

// ---- output projection: logits[(b,l),v] = h2_seq @ Wout^T + b_out ----------
// 4 M-tiles per wave share one B tile (quarters W_out L2 traffic).

__global__ void k_logits(const bf16_t* __restrict__ h2_seq,  // (L*B, 512)
                         const bf16_t* __restrict__ wout,    // (2000, 512)
                         const float*  __restrict__ b_out,   // (2000)
                         float*        __restrict__ out)     // (B, L, 2000)
{
  int wave = (blockIdx.x * blockDim.x + threadIdx.x) >> 5;
  int lane = threadIdx.x & 31;
  const int Ntiles = 125;                  // 2000 / 16 exactly
  if (wave >= 512 * Ntiles) return;        // 2048 M-tiles / 4 per wave
  int mg = wave / Ntiles;                  // M-group: rows mg*64 .. mg*64+63
  int nt = wave % Ntiles;
  v8f acc[4] = {{}, {}, {}, {}};
  for (int k0 = 0; k0 < 512; k0 += 32) {
    if (k0 + 32 < 512)                     // hint: global_prefetch_b8
      __builtin_prefetch(wout + (size_t)(nt * 16 + (lane & 15)) * 512 + k0 + 32, 0, 3);
    v16bf b = load_b_tile(wout, 512, nt * 16, k0, lane);
#pragma unroll
    for (int q = 0; q < 4; ++q) {
      v16bf a = load_a_tile(h2_seq, 512, mg * 64 + q * 16, k0, lane);
      acc[q] = wmma_bf16(a, b, acc[q]);
    }
  }
  int n    = nt * 16 + (lane & 15);
  float bo = b_out[n];
  int moff = (lane < 16) ? 0 : 8;
#pragma unroll
  for (int q = 0; q < 4; ++q) {
#pragma unroll
    for (int r = 0; r < 8; ++r) {
      int rg = mg * 64 + q * 16 + moff + r;  // row = l*64 + b
      int ll = rg >> 6, bb = rg & 63;
      out[((size_t)bb * 512 + ll) * 2000 + n] = acc[q][r] + bo;
    }
  }
}

// ---- in-place log-softmax over V=2000 per (b,l) row ------------------------

__global__ void k_logsoftmax(float* __restrict__ out) {
  __shared__ float red[256];
  float* p = out + (size_t)blockIdx.x * 2000;
  int tid = threadIdx.x;
  float m = -INFINITY;
  for (int v = tid; v < 2000; v += 256) m = fmaxf(m, p[v]);
  red[tid] = m; __syncthreads();
  for (int s = 128; s > 0; s >>= 1) {
    if (tid < s) red[tid] = fmaxf(red[tid], red[tid + s]);
    __syncthreads();
  }
  m = red[0]; __syncthreads();
  float s = 0.0f;
  for (int v = tid; v < 2000; v += 256) s += __expf(p[v] - m);
  red[tid] = s; __syncthreads();
  for (int st = 128; st > 0; st >>= 1) {
    if (tid < st) red[tid] += red[tid + st];
    __syncthreads();
  }
  float lse = m + __logf(red[0]);
  for (int v = tid; v < 2000; v += 256) p[v] -= lse;
}

// ---------------------------------------------------------------------------

extern "C" void kernel_launch(void* const* d_in, const int* in_sizes, int n_in,
                              void* d_out, int out_size, void* d_ws, size_t ws_size,
                              hipStream_t stream) {
  (void)in_sizes; (void)n_in; (void)out_size; (void)ws_size;
  const float* cnn    = (const float*)d_in[0];   // (64,1024)
  const int*   gt     = (const int*)  d_in[1];   // (64,512)
  const float* emb    = (const float*)d_in[2];   // (2000,256)
  const float* wih1   = (const float*)d_in[3];   // (2048,1280)
  const float* whh1   = (const float*)d_in[4];   // (2048,512)
  const float* bih1   = (const float*)d_in[5];
  const float* bhh1   = (const float*)d_in[6];
  const float* wih2   = (const float*)d_in[7];   // (2048,512)
  const float* whh2   = (const float*)d_in[8];
  const float* bih2   = (const float*)d_in[9];
  const float* bhh2   = (const float*)d_in[10];
  const float* wout   = (const float*)d_in[11];  // (2000,512)
  const float* b_out  = (const float*)d_in[12];
  float* out = (float*)d_out;

  // ---- carve workspace (256B aligned chunks) ----
  char* ws = (char*)d_ws;
  auto carve = [&](size_t bytes) -> void* {
    void* p = (void*)ws;
    ws += (bytes + 255) & ~(size_t)255;
    return p;
  };
  bf16_t*   wih1_bf = (bf16_t*)carve((size_t)2048 * 1280 * 2);
  bf16_t*   whh1_bf = (bf16_t*)carve((size_t)2048 * 512 * 2);
  bf16_t*   wih2_bf = (bf16_t*)carve((size_t)2048 * 512 * 2);
  bf16_t*   whh2_bf = (bf16_t*)carve((size_t)2048 * 512 * 2);
  bf16_t*   wout_bf = (bf16_t*)carve((size_t)2000 * 512 * 2);
  bf16_t*   emb_bf  = (bf16_t*)carve((size_t)2000 * 256 * 2);
  bf16_t*   cnn_bf  = (bf16_t*)carve((size_t)64 * 1024 * 2);
  float*    embT    = (float*) carve((size_t)2000 * 2048 * 4);
  float*    base1   = (float*) carve((size_t)64 * 2048 * 4);
  float*    base2   = (float*) carve((size_t)2048 * 4);
  bf16_t*   h1a     = (bf16_t*)carve((size_t)64 * 512 * 2);
  bf16_t*   h1b     = (bf16_t*)carve((size_t)64 * 512 * 2);
  bf16_t*   h2a     = (bf16_t*)carve((size_t)64 * 512 * 2);
  bf16_t*   h2b     = (bf16_t*)carve((size_t)64 * 512 * 2);
  float*    c1      = (float*) carve((size_t)64 * 512 * 4);
  float*    c2      = (float*) carve((size_t)64 * 512 * 4);
  unsigned* bar     = (unsigned*)carve(2 * sizeof(unsigned));
  bf16_t*   h2_seq  = (bf16_t*)carve((size_t)512 * 64 * 512 * 2);

  // ---- precision conversion (weights become L2-resident bf16) ----
  auto conv = [&](const float* src, bf16_t* dst, int n) {
    int blocks = (n + 255) / 256; if (blocks > 2048) blocks = 2048;
    k_f32_to_bf16<<<blocks, 256, 0, stream>>>(src, dst, n);
  };
  conv(wih1, wih1_bf, 2048 * 1280);
  conv(whh1, whh1_bf, 2048 * 512);
  conv(wih2, wih2_bf, 2048 * 512);
  conv(whh2, whh2_bf, 2048 * 512);
  conv(wout, wout_bf, 2000 * 512);
  conv(emb,  emb_bf,  2000 * 256);
  conv(cnn,  cnn_bf,  64 * 1024);

  // ---- zero-init recurrent state + barrier ----
  hipMemsetAsync(h1a, 0, (size_t)64 * 512 * 2, stream);
  hipMemsetAsync(h1b, 0, (size_t)64 * 512 * 2, stream);
  hipMemsetAsync(h2a, 0, (size_t)64 * 512 * 2, stream);
  hipMemsetAsync(h2b, 0, (size_t)64 * 512 * 2, stream);
  hipMemsetAsync(c1,  0, (size_t)64 * 512 * 4, stream);
  hipMemsetAsync(c2,  0, (size_t)64 * 512 * 4, stream);
  hipMemsetAsync(bar, 0, 2 * sizeof(unsigned), stream);

  // base2 = b_ih2 + b_hh2
  k_add_bias<<<8, 256, 0, stream>>>(bih2, bhh2, base2, 2048);

  // embT = embedding @ Wih1[:, 1024:]^T   (M=2000, N=2048, K=256)
  {
    int waves = 125 * 128;
    k_wmma_gemm_bias<<<(waves + 3) / 4, 128, 0, stream>>>(
        emb_bf, 256, wih1_bf + 1024, 1280, embT, 2048,
        125, 128, 256, nullptr, nullptr);
  }
  // base1 = cnn @ Wih1[:, :1024]^T + b_ih1 + b_hh1   (M=64, N=2048, K=1024)
  {
    int waves = 4 * 128;
    k_wmma_gemm_bias<<<(waves + 3) / 4, 128, 0, stream>>>(
        cnn_bf, 1024, wih1_bf, 1280, base1, 2048,
        4, 128, 1024, bih1, bhh1);
  }

  // ---- entire 512-step recurrence: ONE persistent kernel, 512 barriers ----
  k_lstm_persistent<<<32, 128, 0, stream>>>(
      gt, base1, embT, base2, whh1_bf, wih2_bf, whh2_bf,
      c1, c2, h1a, h1b, h2a, h2b, h2_seq, bar, bar + 1);

  // ---- logits GEMM (M=32768, N=2000, K=512) + log-softmax ----
  {
    int waves = 512 * 125;   // 4 M-tiles per wave
    k_logits<<<(waves + 3) / 4, 128, 0, stream>>>(h2_seq, wout_bf, b_out, out);
  }
  k_logsoftmax<<<32768, 256, 0, stream>>>(out);
}
